// DynamicalGraphConv_75084618269081
// MI455X (gfx1250) — matrix operand order repaired and checked by reading, not compiled, not run on hardware
//
#include <hip/hip_runtime.h>
#include <hip/hip_bf16.h>

#define BB    8
#define CC    64
#define NPTS  4096
#define KNN   16
#define OUTC  128
#define EPSBN 1e-5f

typedef __attribute__((ext_vector_type(16))) __bf16 v16bf;
typedef __attribute__((ext_vector_type(8)))  float  v8f;

union Frag  { v16bf v; uint4 u[2]; };
union Pack8 { uint4 u; __bf16 h[8]; };

// ---------------- prep kernels ----------------
__global__ __launch_bounds__(256) void transpose_bf16_kernel(const float* __restrict__ x,
                                                             __bf16* __restrict__ xt) {
    const int gid = blockIdx.x * 256 + threadIdx.x;          // b*C*N + c*N + n
    const int b   = gid / (CC * NPTS);
    const int rem = gid % (CC * NPTS);
    const int c   = rem / NPTS;
    const int n   = rem % NPTS;
    xt[((size_t)(b * NPTS + n)) * CC + c] = (__bf16)x[gid];
}

__global__ __launch_bounds__(256) void sq_kernel(const float* __restrict__ x,
                                                 float* __restrict__ sq) {
    const int gid = blockIdx.x * 256 + threadIdx.x;          // b*N + n
    const int b = gid / NPTS, n = gid % NPTS;
    const float* p = x + (size_t)b * CC * NPTS + n;
    float a = 0.f;
    #pragma unroll 8
    for (int c = 0; c < CC; ++c) { float v = p[(size_t)c * NPTS]; a += v * v; }
    sq[gid] = a;
}

__global__ __launch_bounds__(256) void wconv_kernel(const float* __restrict__ W,
                                                    __bf16* __restrict__ wb) {
    const int gid = blockIdx.x * 256 + threadIdx.x;          // OUTC*2C = 16384
    wb[gid] = (__bf16)W[gid];
}

// ---------------- KNN: pairwise distance (WMMA) + streaming top-16 ----------------
// 128 rows per block, 4 waves; 32-column tile per iteration (8 WMMAs/wave/iter)
__global__ __launch_bounds__(128) void knn_kernel(const __bf16* __restrict__ xt,
                                                  const float* __restrict__ sq,
                                                  int* __restrict__ idx) {
    __shared__ float tile[128][33];
    __shared__ float sqj[32];
    const int b  = blockIdx.x >> 5;            // 32 row-blocks per batch
    const int r0 = (blockIdx.x & 31) << 7;     // 128 rows per block
    const int t  = threadIdx.x;                // 0..127
    const int w  = t >> 5;                     // wave 0..3
    const int l  = t & 31;
    const int lm = l & 15;
    const int khalfA = (l < 16) ? 0 : 8;       // A-matrix ISA layout half
    const int koffB  = (l < 16) ? 0 : 8;       // B-matrix ISA layout half

    const __bf16* xb = xt + (size_t)b * NPTS * CC;

    // hoist A fragments: 2 row-subtiles x 2 K-chunks (rows fixed for whole j-loop)
    Frag afr[2][2];
    #pragma unroll
    for (int s = 0; s < 2; ++s) {
        const int row = r0 + (w << 5) + (s << 4) + lm;
        const __bf16* rp = xb + (size_t)row * CC;
        #pragma unroll
        for (int c = 0; c < 2; ++c) {
            const int c0 = c << 5;
            afr[s][c].u[0] = *(const uint4*)(rp + c0 + khalfA);
            afr[s][c].u[1] = *(const uint4*)(rp + c0 + khalfA + 16);
        }
    }

    const int myrow = r0 + t;
    const float sqi = sq[b * NPTS + myrow];

    float bv[KNN]; int bj[KNN];
    #pragma unroll
    for (int q = 0; q < KNN; ++q) { bv[q] = -3.0e38f; bj[q] = 0; }
    float bmin = -3.0e38f;

    for (int j0 = 0; j0 < NPTS; j0 += 32) {
        // B fragments: 2 col-subtiles x 2 K-chunks
        Frag bfr[2][2];
        #pragma unroll
        for (int jj = 0; jj < 2; ++jj) {
            const __bf16* jp = xb + (size_t)(j0 + (jj << 4) + lm) * CC + koffB;
            #pragma unroll
            for (int c = 0; c < 2; ++c) {
                bfr[jj][c].u[0] = *(const uint4*)(jp + (c << 5));
                bfr[jj][c].u[1] = *(const uint4*)(jp + (c << 5) + 16);
            }
        }
        if (t < 32) sqj[t] = sq[b * NPTS + j0 + t];

        #pragma unroll
        for (int s = 0; s < 2; ++s) {
            const int rbase = (w << 5) + (s << 4) + ((l < 16) ? 0 : 8);
            #pragma unroll
            for (int jj = 0; jj < 2; ++jj) {
                v8f acc = {0.f, 0.f, 0.f, 0.f, 0.f, 0.f, 0.f, 0.f};
                acc = __builtin_amdgcn_wmma_f32_16x16x32_bf16(false, afr[s][0].v, false, bfr[jj][0].v,
                                                              (short)0, acc, false, false);
                acc = __builtin_amdgcn_wmma_f32_16x16x32_bf16(false, afr[s][1].v, false, bfr[jj][1].v,
                                                              (short)0, acc, false, false);
                #pragma unroll
                for (int v = 0; v < 8; ++v)
                    tile[rbase + v][(jj << 4) + lm] = 2.0f * acc[v];
            }
        }
        __syncthreads();

        // selection: thread t owns row t; pd = 2*inner - |xi|^2 - |xj|^2
        #pragma unroll
        for (int col = 0; col < 32; ++col) {
            const float pd = tile[t][col] - sqi - sqj[col];
            if (pd > bmin) {
                bool done = false;
                #pragma unroll
                for (int q = 0; q < KNN; ++q)
                    if (!done && bv[q] == bmin) { bv[q] = pd; bj[q] = j0 + col; done = true; }
                float m = bv[0];
                #pragma unroll
                for (int q = 1; q < KNN; ++q) m = fminf(m, bv[q]);
                bmin = m;
            }
        }
        __syncthreads();
    }
    int* op = idx + ((size_t)(b * NPTS + myrow)) * KNN;
    #pragma unroll
    for (int q = 0; q < KNN; ++q) op[q] = bj[q];
}

// ---------------- edge GEMM (WMMA) + channel sums + per-point min/max over k ----------------
__global__ __launch_bounds__(256) void edge_gemm_kernel(const __bf16* __restrict__ xt,
                                                        const int* __restrict__ idx,
                                                        const __bf16* __restrict__ wm,
                                                        float* __restrict__ ymax,
                                                        float* __restrict__ ymin,
                                                        float* __restrict__ chsum,
                                                        float* __restrict__ chsum2) {
    __shared__ __bf16 edge[128][136];          // 8 points x 16 neighbors, 128 features, padded
    __shared__ float csum[OUTC], csum2[OUTC];
    const int b  = blockIdx.x >> 9;            // 512 blocks per batch
    const int n0 = (blockIdx.x & 511) << 3;    // 8 points per block
    const int t  = threadIdx.x;
    const int w  = t >> 5, l = t & 31, lm = l & 15;

    if (t < OUTC) { csum[t] = 0.f; csum2[t] = 0.f; }

    // build edge tile: row = 16*p + k ; cols [0,64)=nbr-center, [64,128)=center
    {
        const int row = t >> 1, half = t & 1;
        const int p = row >> 4, k = row & 15;
        const int n = n0 + p;
        const __bf16* ct = xt + ((size_t)(b * NPTS + n)) * CC;
        if (half == 0) {
            const int nbr = idx[((size_t)(b * NPTS + n)) * KNN + k];
            const __bf16* nb = xt + ((size_t)(b * NPTS + nbr)) * CC;
            #pragma unroll
            for (int ch = 0; ch < 8; ++ch) {
                Pack8 a, c, r;
                a.u = *(const uint4*)(nb + ch * 8);
                c.u = *(const uint4*)(ct + ch * 8);
                #pragma unroll
                for (int e = 0; e < 8; ++e)
                    r.h[e] = (__bf16)((float)a.h[e] - (float)c.h[e]);
                *(uint4*)(&edge[row][ch * 8]) = r.u;
            }
        } else {
            #pragma unroll
            for (int ch = 0; ch < 8; ++ch)
                *(uint4*)(&edge[row][64 + ch * 8]) = *(const uint4*)(ct + ch * 8);
        }
    }
    __syncthreads();

    const int khalfA = (l < 16) ? 0 : 8;
    const int koffB  = (l < 16) ? 0 : 8;
    v8f acc[8];
    #pragma unroll
    for (int j = 0; j < 8; ++j) acc[j] = (v8f){0.f, 0.f, 0.f, 0.f, 0.f, 0.f, 0.f, 0.f};

    #pragma unroll
    for (int c = 0; c < 4; ++c) {              // K = 128 in chunks of 32
        const int c0 = c << 5;
        Frag af;
        const __bf16* ap = &edge[(w << 4) + lm][c0 + khalfA];
        af.u[0] = *(const uint4*)(ap);
        af.u[1] = *(const uint4*)(ap + 16);
        #pragma unroll
        for (int j = 0; j < 8; ++j) {          // 8 tiles of 16 output channels
            Frag bf_;
            const __bf16* bp = wm + (size_t)((j << 4) + lm) * (2 * CC) + c0 + koffB;
            bf_.u[0] = *(const uint4*)(bp);
            bf_.u[1] = *(const uint4*)(bp + 16);
            acc[j] = __builtin_amdgcn_wmma_f32_16x16x32_bf16(false, af.v, false, bf_.v,
                                                             (short)0, acc[j], false, false);
        }
    }

    // each wave holds point n = n0 + w ; rows of its tile are the 16 k-neighbors
    const int n = n0 + w;
    #pragma unroll
    for (int j = 0; j < 8; ++j) {
        const int o = (j << 4) + lm;
        float s = 0.f, s2 = 0.f, mx = -3.0e38f, mn = 3.0e38f;
        #pragma unroll
        for (int v = 0; v < 8; ++v) {
            const float y = acc[j][v];
            s += y; s2 += y * y;
            mx = fmaxf(mx, y); mn = fminf(mn, y);
        }
        atomicAdd(&csum[o], s);
        atomicAdd(&csum2[o], s2);
        const float omx = __shfl_xor(mx, 16);
        const float omn = __shfl_xor(mn, 16);
        mx = fmaxf(mx, omx); mn = fminf(mn, omn);
        if (l < 16) {
            const size_t oi = ((size_t)(b * OUTC + o)) * NPTS + n;
            ymax[oi] = mx;
            ymin[oi] = mn;
        }
    }
    __syncthreads();
    if (t < OUTC) {
        unsafeAtomicAdd(&chsum[t],  csum[t]);
        unsafeAtomicAdd(&chsum2[t], csum2[t]);
    }
}

// ---------------- BN scale/shift precompute ----------------
__global__ void scale_kernel(const float* __restrict__ chsum, const float* __restrict__ chsum2,
                             const float* __restrict__ gamma, const float* __restrict__ beta,
                             float* __restrict__ sarr, float* __restrict__ tarr) {
    const int o = threadIdx.x;
    const float cnt  = (float)BB * (float)NPTS * (float)KNN;
    const float mean = chsum[o] / cnt;
    const float var  = chsum2[o] / cnt - mean * mean;
    const float s    = gamma[o] * rsqrtf(var + EPSBN);
    sarr[o] = s;
    tarr[o] = beta[o] - mean * s;
}

// ---------------- finalize: affine + relu + max over k via min/max trick ----------------
__global__ __launch_bounds__(256) void finalize_kernel(const float* __restrict__ ymax,
                                                       const float* __restrict__ ymin,
                                                       const float* __restrict__ sarr,
                                                       const float* __restrict__ tarr,
                                                       float* __restrict__ out) {
    const int gid = blockIdx.x * 256 + threadIdx.x;   // (b, o, n) row-major
    const int o = (gid >> 12) & (OUTC - 1);           // NPTS = 4096 = 2^12
    const float s = sarr[o], t = tarr[o];
    float v;
    if (s > 0.f)      v = s * ymax[gid] + t;
    else if (s < 0.f) v = s * ymin[gid] + t;
    else              v = t;
    out[gid] = fmaxf(v, 0.f);
}

extern "C" void kernel_launch(void* const* d_in, const int* in_sizes, int n_in,
                              void* d_out, int out_size, void* d_ws, size_t ws_size,
                              hipStream_t stream) {
    const float* x     = (const float*)d_in[0];
    const float* W     = (const float*)d_in[1];
    const float* gamma = (const float*)d_in[2];
    const float* beta  = (const float*)d_in[3];
    float* out = (float*)d_out;
    char*  ws  = (char*)d_ws;

    __bf16* xt    = (__bf16*)(ws + 0);          //  4,194,304 B  (B,N,C) bf16
    __bf16* wbf   = (__bf16*)(ws + 4194304);    //     32,768 B  (OUT,2C) bf16
    float*  sqv   = (float*) (ws + 4227072);    //    131,072 B  (B,N)
    int*    idxp  = (int*)   (ws + 4358144);    //  2,097,152 B  (B,N,K)
    float*  ymax  = (float*) (ws + 6455296);    // 16,777,216 B  (B,OUT,N)
    float*  ymin  = (float*) (ws + 23232512);   // 16,777,216 B
    float*  chsum = (float*) (ws + 40009728);   //        512 B
    float*  chsum2= (float*) (ws + 40010240);   //        512 B
    float*  sarr  = (float*) (ws + 40010752);   //        512 B
    float*  tarr  = (float*) (ws + 40011264);   //        512 B

    hipMemsetAsync(ws + 40009728, 0, 1024, stream);   // zero chsum/chsum2 each call

    transpose_bf16_kernel<<<(BB * CC * NPTS) / 256, 256, 0, stream>>>(x, xt);
    sq_kernel<<<(BB * NPTS) / 256, 256, 0, stream>>>(x, sqv);
    wconv_kernel<<<(OUTC * 2 * CC) / 256, 256, 0, stream>>>(W, wbf);
    knn_kernel<<<BB * (NPTS / 128), 128, 0, stream>>>(xt, sqv, idxp);
    edge_gemm_kernel<<<BB * (NPTS / 8), 256, 0, stream>>>(xt, idxp, wbf, ymax, ymin, chsum, chsum2);
    scale_kernel<<<1, OUTC, 0, stream>>>(chsum, chsum2, gamma, beta, sarr, tarr);
    finalize_kernel<<<(BB * OUTC * NPTS) / 256, 256, 0, stream>>>(ymax, ymin, sarr, tarr, out);
}
